// VariationalLinearEncoder_22308060136297
// MI455X (gfx1250) — matrix operand order repaired and checked by reading, not compiled, not run on hardware
//
#include <hip/hip_runtime.h>
#include <hip/hip_bf16.h>

typedef __attribute__((ext_vector_type(16))) __bf16 v16bf;
typedef __attribute__((ext_vector_type(8)))  float  v8f;

#define N_NODES 50000
#define N_EDGES 800000
#define IN_CH   512
#define OUT_CH  128
#define K_STEPS (IN_CH / 32)    // 16 k-steps of 32
#define N_TILES (OUT_CH / 16)   // 8 column tiles
#define M_TILES (N_NODES / 16)  // 3125 row tiles (exact)
#define FRAG_ELEMS 512          // 32 lanes * 16 bf16 per B fragment
#define WPK_ELEMS (2 * K_STEPS * N_TILES * FRAG_ELEMS)   // 131072 bf16
#define WPK_BYTES (WPK_ELEMS * 2)                        // 262144 B (fits 320KB LDS)

// ---------------- degree ----------------
__global__ void k_deg_init(float* __restrict__ deg) {
    int i = blockIdx.x * 256 + threadIdx.x;
    if (i < N_NODES) deg[i] = 1.0f;   // self-loop
}

__global__ void k_deg(const int* __restrict__ ei, float* __restrict__ deg) {
    int e = blockIdx.x * 256 + threadIdx.x;
    if (e < N_EDGES) atomicAdd(&deg[ei[N_EDGES + e]], 1.0f);  // dst degree
}

// ------------- repack W (f32 [512,128]) into bf16 WMMA B-fragment layout -------------
// flat index = (((m*16+s)*8+n)*32 + lane)*16 + j
// lane l: col = n*16 + (l&15); K = s*32 + (l>>4)*8 + (j&7) + (j>>3)*16   (16-bit B 32x16 layout)
__global__ void k_pack_w(const float* __restrict__ Wm, const float* __restrict__ Wl,
                         __bf16* __restrict__ wpk) {
    int tid = blockIdx.x * 256 + threadIdx.x;             // 0 .. 131071
    if (tid >= WPK_ELEMS) return;
    int j    = tid & 15;
    int lane = (tid >> 4) & 31;
    int n    = (tid >> 9) & 7;
    int s    = (tid >> 12) & 15;
    int m    = tid >> 16;
    const float* W = m ? Wl : Wm;
    int k = s * 32 + ((lane >> 4) << 3) + (j & 7) + ((j >> 3) << 4);
    int c = n * 16 + (lane & 15);
    wpk[tid] = (__bf16)W[k * OUT_CH + c];
}

// ------------- WMMA GEMM: h_mu = x@W_mu, h_ls = x@W_logstd -------------
// 8 waves/block, one 16-row tile per wave, both weights per wave (x read once).
// B fragments served from LDS (staged once per block); A prefetched one k-step ahead.
__global__ void __launch_bounds__(256)
k_gemm(const float* __restrict__ x, const __bf16* __restrict__ wpk,
       float* __restrict__ hmu, float* __restrict__ hls) {
    extern __shared__ char smemraw[];

    // stage packed W (both matrices, 256KB) into LDS
    {
        uint4* sdst = (uint4*)smemraw;
        const uint4* ssrc = (const uint4*)wpk;
        for (int i = threadIdx.x; i < WPK_BYTES / 16; i += 256) sdst[i] = ssrc[i];
    }
    __syncthreads();

    int wave = blockIdx.x * 8 + (threadIdx.x >> 5);
    if (wave >= M_TILES) return;                 // uniform per wave; EXEC stays all-1s
    int lane  = threadIdx.x & 31;
    int khalf = lane >> 4;
    int row   = wave * 16 + (lane & 15);

    const __bf16* lwm = (const __bf16*)smemraw;            // mu fragments
    const __bf16* lwl = lwm + (WPK_ELEMS / 2);             // logstd fragments

    v8f accm[8], accl[8];
    v8f zero = {};
#pragma unroll
    for (int n = 0; n < 8; ++n) { accm[n] = zero; accl[n] = zero; }

    const float* xrow = x + (size_t)row * IN_CH;

    // preload A for s = 0
    int k0 = khalf * 8;
    float4 f0 = *(const float4*)(xrow + k0);
    float4 f1 = *(const float4*)(xrow + k0 + 4);
    float4 f2 = *(const float4*)(xrow + k0 + 16);
    float4 f3 = *(const float4*)(xrow + k0 + 20);

    for (int s = 0; s < K_STEPS; ++s) {
        // convert current A tile: lane<16 holds K {k..k+7, k+16..k+23}, lanes>=16 shifted by 8
        v16bf a;
        a[0]=(__bf16)f0.x;  a[1]=(__bf16)f0.y;  a[2]=(__bf16)f0.z;  a[3]=(__bf16)f0.w;
        a[4]=(__bf16)f1.x;  a[5]=(__bf16)f1.y;  a[6]=(__bf16)f1.z;  a[7]=(__bf16)f1.w;
        a[8]=(__bf16)f2.x;  a[9]=(__bf16)f2.y;  a[10]=(__bf16)f2.z; a[11]=(__bf16)f2.w;
        a[12]=(__bf16)f3.x; a[13]=(__bf16)f3.y; a[14]=(__bf16)f3.z; a[15]=(__bf16)f3.w;

        // prefetch next k-step's A while WMMAs run
        if (s + 1 < K_STEPS) {
            int k = (s + 1) * 32 + khalf * 8;
            f0 = *(const float4*)(xrow + k);
            f1 = *(const float4*)(xrow + k + 4);
            f2 = *(const float4*)(xrow + k + 16);
            f3 = *(const float4*)(xrow + k + 20);
        }

#pragma unroll
        for (int n = 0; n < 8; ++n) {
            v16bf bm = *(const v16bf*)(lwm + ((s * 8 + n) * 32 + lane) * 16);
            accm[n] = __builtin_amdgcn_wmma_f32_16x16x32_bf16(
                false, a, false, bm, (short)0, accm[n], false, false);
            v16bf bl = *(const v16bf*)(lwl + ((s * 8 + n) * 32 + lane) * 16);
            accl[n] = __builtin_amdgcn_wmma_f32_16x16x32_bf16(
                false, a, false, bl, (short)0, accl[n], false, false);
        }
    }

    // C/D layout: VGPR v, lanes 0-15 -> M=v, N=lane; lanes 16-31 -> M=v+8, N=lane-16
    int col   = lane & 15;
    int rbase = wave * 16 + (khalf << 3);
#pragma unroll
    for (int n = 0; n < 8; ++n)
#pragma unroll
        for (int v = 0; v < 8; ++v) {
            hmu[(size_t)(rbase + v) * OUT_CH + n * 16 + col] = accm[n][v];
            hls[(size_t)(rbase + v) * OUT_CH + n * 16 + col] = accl[n][v];
        }
}

// ------------- self loops: out = h*dinv^2 + b ; also materialize dinv -------------
__global__ void k_self(const float* __restrict__ hmu, const float* __restrict__ hls,
                       float* __restrict__ deg_dinv,
                       const float* __restrict__ bmu, const float* __restrict__ bls,
                       float* __restrict__ omu, float* __restrict__ ols) {
    int tid = blockIdx.x * 256 + threadIdx.x;   // one warp per row, 4 cols per lane
    int r = tid >> 5;
    if (r >= N_NODES) return;
    int c = (tid & 31) << 2;
    float d  = rsqrtf(deg_dinv[r]);
    float n2 = d * d;
    if ((tid & 31) == 0) deg_dinv[r] = d;       // in-warp read-before-write: safe
    float4 hm = *(const float4*)(hmu + (size_t)r * OUT_CH + c);
    float4 hl = *(const float4*)(hls + (size_t)r * OUT_CH + c);
    float4 b0 = *(const float4*)(bmu + c);
    float4 b1 = *(const float4*)(bls + c);
    float4 om, ol;
    om.x = hm.x * n2 + b0.x; om.y = hm.y * n2 + b0.y;
    om.z = hm.z * n2 + b0.z; om.w = hm.w * n2 + b0.w;
    ol.x = hl.x * n2 + b1.x; ol.y = hl.y * n2 + b1.y;
    ol.z = hl.z * n2 + b1.z; ol.w = hl.w * n2 + b1.w;
    *(float4*)(omu + (size_t)r * OUT_CH + c) = om;
    *(float4*)(ols + (size_t)r * OUT_CH + c) = ol;
}

// ------------- edge aggregation: out[dst] += h[src] * dinv[src]*dinv[dst] -------------
__global__ void k_edge(const int* __restrict__ ei, const float* __restrict__ dinv,
                       const float* __restrict__ hmu, const float* __restrict__ hls,
                       float* __restrict__ omu, float* __restrict__ ols) {
    int lane = threadIdx.x & 31;
    int wid  = blockIdx.x * (blockDim.x >> 5) + (threadIdx.x >> 5);
    int nw   = gridDim.x * (blockDim.x >> 5);
    int c    = lane << 2;                         // 32 lanes x float4 = 128 cols
    for (int e = wid; e < N_EDGES; e += nw) {
        int s = ei[e];
        int d = ei[N_EDGES + e];
        float norm = dinv[s] * dinv[d];
        float4 hm = *(const float4*)(hmu + (size_t)s * OUT_CH + c);
        float4 hl = *(const float4*)(hls + (size_t)s * OUT_CH + c);
        float* om = omu + (size_t)d * OUT_CH + c;
        atomicAdd(om + 0, hm.x * norm); atomicAdd(om + 1, hm.y * norm);
        atomicAdd(om + 2, hm.z * norm); atomicAdd(om + 3, hm.w * norm);
        float* ol = ols + (size_t)d * OUT_CH + c;
        atomicAdd(ol + 0, hl.x * norm); atomicAdd(ol + 1, hl.y * norm);
        atomicAdd(ol + 2, hl.z * norm); atomicAdd(ol + 3, hl.w * norm);
    }
}

extern "C" void kernel_launch(void* const* d_in, const int* in_sizes, int n_in,
                              void* d_out, int out_size, void* d_ws, size_t ws_size,
                              hipStream_t stream) {
    const float* x   = (const float*)d_in[0];
    const int*   ei  = (const int*)d_in[1];      // [2, E] (src row, dst row)
    const float* Wm  = (const float*)d_in[2];
    const float* bmu = (const float*)d_in[3];
    const float* Wl  = (const float*)d_in[4];
    const float* bls = (const float*)d_in[5];

    // workspace layout (floats): [deg/dinv 50176-pad][h_mu 6.4M][h_ls 6.4M][wpk bf16 256KB]
    float*  ws   = (float*)d_ws;
    float*  deg  = ws;
    float*  hmu  = ws + 50176;                   // 256B aligned
    float*  hls  = hmu + (size_t)N_NODES * OUT_CH;
    __bf16* wpk  = (__bf16*)(hls + (size_t)N_NODES * OUT_CH);

    float* omu = (float*)d_out;
    float* ols = omu + (size_t)N_NODES * OUT_CH;

    k_deg_init<<<(N_NODES + 255) / 256, 256, 0, stream>>>(deg);
    k_deg<<<(N_EDGES + 255) / 256, 256, 0, stream>>>(ei, deg);
    k_pack_w<<<512, 256, 0, stream>>>(Wm, Wl, wpk);
    k_gemm<<<(M_TILES + 7) / 8, 256, WPK_BYTES, stream>>>(x, wpk, hmu, hls);
    k_self<<<(N_NODES * 32 + 255) / 256, 256, 0, stream>>>(hmu, hls, deg, bmu, bls, omu, ols);
    k_edge<<<4096, 256, 0, stream>>>(ei, deg, hmu, hls, omu, ols);
}
